// Model_14336600834737
// MI455X (gfx1250) — matrix operand order repaired and checked by reading, not compiled, not run on hardware
//
#include <hip/hip_runtime.h>
#include <hip/hip_bf16.h>
#include <math.h>

typedef __bf16 bf16;
typedef __attribute__((ext_vector_type(16))) __bf16 v16bf;
typedef __attribute__((ext_vector_type(8)))  float  v8f;

namespace {
constexpr int kD   = 384;     // model dim
constexpr int kM   = 16384;   // B*N tokens
constexpr int kN   = 2048;    // sequence length per batch
constexpr int kEX  = 768;
constexpr int kTS  = 128;     // qhkh token stride: qh[0..63] (48 real) + kh[64..127]
constexpr float kScale = 0.14433756729740643f; // 1/sqrt(48)
}

struct B32 { int4 a, b; };

// load a 32-byte WMMA fragment as two 16-byte chunks
__device__ __forceinline__ v16bf ld2(const bf16* p0, const bf16* p1) {
  B32 t; t.a = *(const int4*)p0; t.b = *(const int4*)p1;
  return __builtin_bit_cast(v16bf, t);
}

// A fragment (16x32 bf16) from row-major [row, k], leading dim lda.
// Lane l: row = row0 + (l&15); elems = k0 + 16*(l>>4) + 0..15 (contiguous).
__device__ __forceinline__ v16bf fragA(const bf16* A, int lda, int row0, int k0, int lane) {
  const bf16* p = A + (size_t)(row0 + (lane & 15)) * lda + k0 + 16 * (lane >> 4);
  return ld2(p, p + 8);
}

// B fragment (32x16 bf16) where W is row-major [outcol, k], leading dim ldw.
// Lane l: col = col0 + (l&15); elems = k0 + 16*(l>>4) + 0..15 (contiguous).
__device__ __forceinline__ v16bf fragB(const bf16* W, int ldw, int col0, int k0, int lane) {
  const bf16* p = W + (size_t)(col0 + (lane & 15)) * ldw + k0 + 16 * (lane >> 4);
  return ld2(p, p + 8);
}

__device__ __forceinline__ v8f wmma_bf16(v16bf a, v16bf b, v8f c) {
  return __builtin_amdgcn_wmma_f32_16x16x32_bf16(false, a, false, b, (short)0, c, false, false);
}

// Fully-unrolled, scheduler-pinned software pipeline.
// NA = 16-row A fragments per wave (32-row tile for NA=2), NF = 16-col B frags.
// sched_barrier(0) forbids the machine scheduler from sinking step s+1's
// loads below step s's WMMA chain, so WMMA waits keep a full K-step of loads
// in flight instead of draining loadcnt to 0.
template <int NA, int NF, int K>
__device__ __forceinline__ void gemm_pipe(const bf16* __restrict__ A, int lda, int row0,
    const bf16* __restrict__ W, int ldw, int col0, int lane, v8f (&acc)[NA][NF]) {
  constexpr int S = K / 32;
  v16bf a[2][NA];
  v16bf b[2][NF];
#pragma unroll
  for (int r = 0; r < NA; ++r) a[0][r] = fragA(A, lda, row0 + 16 * r, 0, lane);
#pragma unroll
  for (int c = 0; c < NF; ++c) b[0][c] = fragB(W, ldw, col0 + 16 * c, 0, lane);
#pragma unroll
  for (int s = 0; s < S; ++s) {
    const int cur = s & 1, nxt = cur ^ 1;
    if (s + 1 < S) {
#pragma unroll
      for (int r = 0; r < NA; ++r)
        a[nxt][r] = fragA(A, lda, row0 + 16 * r, (s + 1) * 32, lane);
#pragma unroll
      for (int c = 0; c < NF; ++c)
        b[nxt][c] = fragB(W, ldw, col0 + 16 * c, (s + 1) * 32, lane);
    }
    __builtin_amdgcn_sched_barrier(0);   // loads(s+1) stay above wmma(s)
#pragma unroll
    for (int r = 0; r < NA; ++r)
#pragma unroll
      for (int c = 0; c < NF; ++c)
        acc[r][c] = wmma_bf16(a[cur][r], b[cur][c], acc[r][c]);
    __builtin_amdgcn_sched_barrier(0);
  }
}

// Dual-weight pinned pipeline (shared A, two weight bases) for fused GEGLU.
template <int NA, int NF, int K>
__device__ __forceinline__ void gemm_pipe2(const bf16* __restrict__ A, int lda, int row0,
    const bf16* __restrict__ W1, const bf16* __restrict__ W2, int ldw, int lane,
    v8f (&accL)[NA][NF], v8f (&accP)[NA][NF]) {
  constexpr int S = K / 32;
  v16bf a[2][NA];
  v16bf b1[2][NF], b2[2][NF];
#pragma unroll
  for (int r = 0; r < NA; ++r) a[0][r] = fragA(A, lda, row0 + 16 * r, 0, lane);
#pragma unroll
  for (int c = 0; c < NF; ++c) {
    b1[0][c] = fragB(W1, ldw, 16 * c, 0, lane);
    b2[0][c] = fragB(W2, ldw, 16 * c, 0, lane);
  }
#pragma unroll
  for (int s = 0; s < S; ++s) {
    const int cur = s & 1, nxt = cur ^ 1;
    if (s + 1 < S) {
#pragma unroll
      for (int r = 0; r < NA; ++r)
        a[nxt][r] = fragA(A, lda, row0 + 16 * r, (s + 1) * 32, lane);
#pragma unroll
      for (int c = 0; c < NF; ++c) {
        b1[nxt][c] = fragB(W1, ldw, 16 * c, (s + 1) * 32, lane);
        b2[nxt][c] = fragB(W2, ldw, 16 * c, (s + 1) * 32, lane);
      }
    }
    __builtin_amdgcn_sched_barrier(0);
#pragma unroll
    for (int r = 0; r < NA; ++r)
#pragma unroll
      for (int c = 0; c < NF; ++c) {
        accL[r][c] = wmma_bf16(a[cur][r], b1[cur][c], accL[r][c]);
        accP[r][c] = wmma_bf16(a[cur][r], b2[cur][c], accP[r][c]);
      }
    __builtin_amdgcn_sched_barrier(0);
  }
}

__device__ __forceinline__ float redmax16(float v) {
  v = fmaxf(v, __shfl_xor(v, 1, 32));
  v = fmaxf(v, __shfl_xor(v, 2, 32));
  v = fmaxf(v, __shfl_xor(v, 4, 32));
  v = fmaxf(v, __shfl_xor(v, 8, 32));
  return v;
}
__device__ __forceinline__ float redsum16(float v) {
  v += __shfl_xor(v, 1, 32);
  v += __shfl_xor(v, 2, 32);
  v += __shfl_xor(v, 4, 32);
  v += __shfl_xor(v, 8, 32);
  return v;
}
__device__ __forceinline__ float redsum32(float v) {
  v = redsum16(v);
  v += __shfl_xor(v, 16, 32);
  return v;
}

// ---------------- utility kernels ----------------
__global__ __launch_bounds__(256) void k_cvt(const float* __restrict__ s,
                                             bf16* __restrict__ d, int n) {
  int i = blockIdx.x * 256 + threadIdx.x;
  if (i < n) d[i] = (bf16)s[i];
}

__global__ __launch_bounds__(256) void k_zero(int4* __restrict__ p, int n) {
  int i = blockIdx.x * 256 + threadIdx.x;
  if (i < n) { int4 z; z.x = z.y = z.z = z.w = 0; p[i] = z; }
}

// ---------------- embedding + first layernorm ----------------
__global__ __launch_bounds__(256) void k_embed(const float* __restrict__ q,
    const float* __restrict__ gemb, const float* __restrict__ g,
    const float* __restrict__ b, float* __restrict__ x) {
  int lane = threadIdx.x & 31, wave = threadIdx.x >> 5;
  int m = blockIdx.x * 8 + wave;
  float qv = q[m];
  int grp = m & 1;   // GS=2, N even -> group = token%2
  float v[12]; float s = 0.f;
#pragma unroll
  for (int i = 0; i < 12; ++i) {
    int d = lane + 32 * i;
    int f = (d < 192) ? d : d - 192;
    float fr = __expf((float)f * -0.04797052465948999f); // exp(2f * -ln(1e4)/384)
    float e = qv * fr;
    float val = (d < 192) ? __sinf(e) : __cosf(e);
    val += gemb[grp * kD + d];
    v[i] = val; s += val;
  }
  float mu = redsum32(s) * (1.f / kD);
  float var = 0.f;
#pragma unroll
  for (int i = 0; i < 12; ++i) { float dd = v[i] - mu; var += dd * dd; }
  var = redsum32(var) * (1.f / kD);
  float inv = rsqrtf(var + 1e-5f);
  float* xr = x + (size_t)m * kD;
#pragma unroll
  for (int i = 0; i < 12; ++i) {
    int d = lane + 32 * i;
    xr[d] = (v[i] - mu) * inv * g[d] + b[d];
  }
}

// ---------------- layernorm (f32 in -> bf16 out) ----------------
__global__ __launch_bounds__(256) void k_ln(const float* __restrict__ x,
    const float* __restrict__ g, const float* __restrict__ b,
    bf16* __restrict__ h) {
  int lane = threadIdx.x & 31, wave = threadIdx.x >> 5;
  int m = blockIdx.x * 8 + wave;
  const float* xr = x + (size_t)m * kD;
  float v[12]; float s = 0.f;
#pragma unroll
  for (int i = 0; i < 12; ++i) { v[i] = xr[lane + 32 * i]; s += v[i]; }
  float mu = redsum32(s) * (1.f / kD);
  float var = 0.f;
#pragma unroll
  for (int i = 0; i < 12; ++i) { float dd = v[i] - mu; var += dd * dd; }
  var = redsum32(var) * (1.f / kD);
  float inv = rsqrtf(var + 1e-5f);
  bf16* hr = h + (size_t)m * kD;
#pragma unroll
  for (int i = 0; i < 12; ++i) {
    int d = lane + 32 * i;
    hr[d] = (bf16)((v[i] - mu) * inv * g[d] + b[d]);
  }
}

// ---------------- expand GEMM, qh/kh part (cols 0..95) ----------------
// grid (M/128, 2): 32-row wave tiles, col tiles of 48. Writes padded
// qhkh[token*128 + pos], pos in {0..47, 64..111}; pad pre-zeroed.
__global__ __launch_bounds__(128) void k_gemm_qk(const bf16* __restrict__ h,
    const bf16* __restrict__ W, bf16* __restrict__ qhkh) {
  int lane = threadIdx.x & 31, wave = threadIdx.x >> 5;
  int row0 = (blockIdx.x * 4 + wave) * 32;
  int n0 = blockIdx.y * 48;
  v8f acc[2][3] = {};
  gemm_pipe<2, 3, kD>(h, kD, row0, W, kD, n0, lane, acc);
  int nn = lane & 15, half = lane >> 4;
#pragma unroll
  for (int r = 0; r < 2; ++r)
#pragma unroll
    for (int c = 0; c < 3; ++c) {
      int n = n0 + c * 16 + nn;
      int pos = (n < 48) ? n : 64 + (n - 48);
#pragma unroll
      for (int j = 0; j < 8; ++j)
        qhkh[(size_t)(row0 + 16 * r + j + 8 * half) * kTS + pos] = (bf16)acc[r][c][j];
    }
}

// ---------------- expand GEMM, GEGLU part (fused lin/pre_gelu cols) --------
// grid (M/128, 24): 32-row wave tiles, col tiles of 32.
// geglu col n: lin = z[:,96+n], pre = z[:,864+n]; out = lin*gelu(pre)
// n<384 -> cat[:, n] (local);  n>=384 -> attnvT[n-384][token]
__global__ __launch_bounds__(128) void k_gemm_geglu(const bf16* __restrict__ h,
    const bf16* __restrict__ W, bf16* __restrict__ cat, bf16* __restrict__ attnvT) {
  int lane = threadIdx.x & 31, wave = threadIdx.x >> 5;
  int row0 = (blockIdx.x * 4 + wave) * 32;
  int n0 = blockIdx.y * 32;
  const bf16* W1 = W + (size_t)(96 + n0) * kD;
  const bf16* W2 = W + (size_t)(864 + n0) * kD;
  v8f accL[2][2] = {}, accP[2][2] = {};
  gemm_pipe2<2, 2, kD>(h, kD, row0, W1, W2, kD, lane, accL, accP);
  int nn = lane & 15, half = lane >> 4;
#pragma unroll
  for (int r = 0; r < 2; ++r)
#pragma unroll
    for (int c = 0; c < 2; ++c) {
      int n = n0 + c * 16 + nn;
#pragma unroll
      for (int j = 0; j < 8; ++j) {
        int m = row0 + 16 * r + j + 8 * half;
        float pre = accP[r][c][j];
        float gg = accL[r][c][j] * 0.5f * pre * (1.f + erff(pre * 0.7071067811865476f));
        if (n < 384) cat[(size_t)m * kEX + n] = (bf16)gg;
        else         attnvT[(size_t)(n - 384) * kM + m] = (bf16)gg;
      }
    }
}

// ---------------- flash attention ----------------
// grid.x = query tiles (16 rows); 4 waves, each owns 96 V columns.
// Manual unroll-by-2 with two named K-fragment buffers + sched barriers so
// the prefetch distance of one 32-key tile survives scheduling.
__global__ __launch_bounds__(128) void k_attn(const bf16* __restrict__ qhkh,
    const bf16* __restrict__ attnvT, const float* __restrict__ pbm_l,
    bf16* __restrict__ cat) {
  __shared__ __align__(16) bf16 pshare[4][16 * 32];
  int lane = threadIdx.x & 31, wave = threadIdx.x >> 5;
  int nn = lane & 15, half = lane >> 4;
  int q0 = blockIdx.x * 16;          // global token
  int batch = q0 >> 11;              // /2048
  int qloc0 = q0 & 2047;
  float bias = pbm_l[0];
  const bf16* khp = qhkh + 64;

  v16bf qa0 = fragA(qhkh, kTS, q0, 0,  lane);
  v16bf qa1 = fragA(qhkh, kTS, q0, 32, lane);   // k 48..63 are zero-padded
  v8f acc[6] = {};
  float mrow[8], lrow[8];
#pragma unroll
  for (int j = 0; j < 8; ++j) { mrow[j] = -INFINITY; lrow[j] = 0.f; }
  int col0 = wave * 96;
  int kend = qloc0 + 16;
  int kg0 = batch * kN;

  auto loadK = [&](v16bf (&kb)[4], int kg) {
    kb[0] = fragB(khp, kTS, kg,      0,  lane);
    kb[1] = fragB(khp, kTS, kg,      32, lane);
    kb[2] = fragB(khp, kTS, kg + 16, 0,  lane);
    kb[3] = fragB(khp, kTS, kg + 16, 32, lane);
    __builtin_amdgcn_sched_barrier(0);   // keep prefetch ahead of consumers
  };

  // one 32-key step: S = QK^T, online softmax, P@V
  auto attn_tile = [&](int kt, const v16bf (&kb)[4]) {
    int kg = kg0 + kt;
    v8f s0 = {}, s1 = {};
    s0 = wmma_bf16(qa0, kb[0], s0);
    s0 = wmma_bf16(qa1, kb[1], s0);
    s1 = wmma_bf16(qa0, kb[2], s1);
    s1 = wmma_bf16(qa1, kb[3], s1);

    int key0 = kt + nn, key1 = kt + 16 + nn;
#pragma unroll
    for (int j = 0; j < 8; ++j) {
      int qrow = qloc0 + j + 8 * half;
      float g0 = 1.f / (1.f + __expf(-((float)(key0 - qrow) + bias)));
      float g1 = 1.f / (1.f + __expf(-((float)(key1 - qrow) + bias)));
      s0[j] = (key0 <= qrow) ? s0[j] * kScale + g0 : -INFINITY;
      s1[j] = (key1 <= qrow) ? s1[j] * kScale + g1 : -INFINITY;
    }
#pragma unroll
    for (int j = 0; j < 8; ++j) {
      float t = redmax16(fmaxf(s0[j], s1[j]));
      float mn = fmaxf(mrow[j], t);
      float corr = __expf(mrow[j] - mn);
      mrow[j] = mn;
      float p0 = __expf(s0[j] - mn), p1 = __expf(s1[j] - mn);
      s0[j] = p0; s1[j] = p1;
      lrow[j] = lrow[j] * corr + redsum16(p0 + p1);
#pragma unroll
      for (int c = 0; c < 6; ++c) acc[c][j] *= corr;
    }
    // V fragments issued before the LDS round trip so they overlap it
    v16bf vb[6];
#pragma unroll
    for (int c = 0; c < 6; ++c) {
      const bf16* vp = attnvT + (size_t)(col0 + c * 16 + nn) * kM + kg + 16 * half;
      vb[c] = ld2(vp, vp + 8);
    }
    __builtin_amdgcn_sched_barrier(0);   // V loads stay above the LDS round trip
    // stage P (16x32) to LDS in [row][key] order, then reload as A fragment
    bf16* pw = &pshare[wave][0];
#pragma unroll
    for (int j = 0; j < 8; ++j) {
      int row = j + 8 * half;
      pw[row * 32 + nn]      = (bf16)s0[j];
      pw[row * 32 + 16 + nn] = (bf16)s1[j];
    }
    __builtin_amdgcn_wave_barrier();   // same-wave LDS is in-order; keep sched order
    const bf16* pr = &pshare[wave][nn * 32 + 16 * half];
    v16bf pA = ld2(pr, pr + 8);
#pragma unroll
    for (int c = 0; c < 6; ++c) acc[c] = wmma_bf16(pA, vb[c], acc[c]);
    __builtin_amdgcn_wave_barrier();
  };

  v16bf kbA[4], kbB[4];
  loadK(kbA, kg0);
  for (int kt = 0; kt < kend; kt += 64) {
    bool has2 = (kt + 32 < kend);
    if (has2) loadK(kbB, kg0 + kt + 32);
    attn_tile(kt, kbA);
    if (!has2) break;
    if (kt + 64 < kend) loadK(kbA, kg0 + kt + 64);
    attn_tile(kt + 32, kbB);
  }

#pragma unroll
  for (int j = 0; j < 8; ++j) {
    float inv = 1.f / lrow[j];
    int m = q0 + j + 8 * half;
#pragma unroll
    for (int c = 0; c < 6; ++c)
      cat[(size_t)m * kEX + 384 + col0 + c * 16 + nn] = (bf16)(acc[c][j] * inv);
  }
}

// ---------------- projection GEMM + residual add ----------------
// grid (M/128, 6): 32-row wave tiles, col tiles of 64.
__global__ __launch_bounds__(128) void k_gemm_proj(const bf16* __restrict__ cat,
    const bf16* __restrict__ W, float* __restrict__ x) {
  int lane = threadIdx.x & 31, wave = threadIdx.x >> 5;
  int row0 = (blockIdx.x * 4 + wave) * 32;
  int n0 = blockIdx.y * 64;
  v8f acc[2][4] = {};
  gemm_pipe<2, 4, kEX>(cat, kEX, row0, W, kEX, n0, lane, acc);
  int nn = lane & 15, half = lane >> 4;
#pragma unroll
  for (int r = 0; r < 2; ++r)
#pragma unroll
    for (int c = 0; c < 4; ++c) {
      int n = n0 + c * 16 + nn;
#pragma unroll
      for (int j = 0; j < 8; ++j)
        x[(size_t)(row0 + 16 * r + j + 8 * half) * kD + n] += acc[r][c][j];
    }
}

// ---------------- final logits GEMM ----------------
__global__ __launch_bounds__(128) void k_gemm_logits(const bf16* __restrict__ h,
    const bf16* __restrict__ W, float* __restrict__ out) {
  int lane = threadIdx.x & 31, wave = threadIdx.x >> 5;
  int row0 = (blockIdx.x * 4 + wave) * 16;
  int n0 = blockIdx.y * 64;
  v8f acc[4] = {};
#pragma unroll
  for (int k = 0; k < kD; k += 32) {
    v16bf a = fragA(h, kD, row0, k, lane);
    v16bf b[4];
#pragma unroll
    for (int c = 0; c < 4; ++c) {
      int col = n0 + c * 16 + (lane & 15);
      if (col > 99) col = 99;               // clamp: keep loads in-bounds
      const bf16* p = W + (size_t)col * kD + k + 16 * (lane >> 4);
      b[c] = ld2(p, p + 8);
    }
#pragma unroll
    for (int c = 0; c < 4; ++c) acc[c] = wmma_bf16(a, b[c], acc[c]);
  }
  int nn = lane & 15, half = lane >> 4;
#pragma unroll
  for (int c = 0; c < 4; ++c) {
    int n = n0 + c * 16 + nn;
    if (n < 100) {
#pragma unroll
      for (int j = 0; j < 8; ++j)
        out[(size_t)(row0 + j + 8 * half) * 100 + n] = acc[c][j];
    }
  }
}

extern "C" void kernel_launch(void* const* d_in, const int* in_sizes, int n_in,
                              void* d_out, int out_size, void* d_ws, size_t ws_size,
                              hipStream_t stream) {
  (void)in_sizes; (void)n_in; (void)out_size; (void)ws_size;
  const float* q     = (const float*)d_in[0];
  const float* gemb  = (const float*)d_in[1];
  const float* expw  = (const float*)d_in[2];
  const float* projw = (const float*)d_in[3];
  const float* lng   = (const float*)d_in[4];
  const float* lnb   = (const float*)d_in[5];
  const float* pbm   = (const float*)d_in[6];
  const float* ng    = (const float*)d_in[7];
  const float* nb    = (const float*)d_in[8];
  const float* outw  = (const float*)d_in[9];
  float* out = (float*)d_out;

  char* p = (char*)d_ws;
  auto take = [&p](size_t bytes) { void* r = (void*)p; p += (bytes + 255) & ~(size_t)255; return r; };
  float* x      = (float*)take((size_t)kM * kD * 4);           // residual stream
  bf16*  h      = (bf16*) take((size_t)kM * kD * 2);           // LN output
  bf16*  qhkh   = (bf16*) take((size_t)kM * kTS * 2);          // padded q/k heads
  bf16*  cat    = (bf16*) take((size_t)kM * kEX * 2);          // [local | attention]
  bf16*  attnvT = (bf16*) take((size_t)kD * kM * 2);           // V transposed [col][tok]
  bf16*  expb   = (bf16*) take((size_t)8 * 1632 * kD * 2);
  bf16*  projb  = (bf16*) take((size_t)8 * kD * kEX * 2);
  bf16*  outwb  = (bf16*) take((size_t)100 * kD * 2);

  int ne = 8 * 1632 * kD;
  k_cvt<<<(ne + 255) / 256, 256, 0, stream>>>(expw, expb, ne);
  int np = 8 * kD * kEX;
  k_cvt<<<(np + 255) / 256, 256, 0, stream>>>(projw, projb, np);
  int no = 100 * kD;
  k_cvt<<<(no + 255) / 256, 256, 0, stream>>>(outw, outwb, no);
  int nz = kM * kTS * 2 / 16;
  k_zero<<<(nz + 255) / 256, 256, 0, stream>>>((int4*)qhkh, nz);  // zero qk padding

  k_embed<<<kM / 8, 256, 0, stream>>>(q, gemb, ng, nb, x);

  for (int l = 0; l < 8; ++l) {
    k_ln<<<kM / 8, 256, 0, stream>>>(x, lng + l * kD, lnb + l * kD, h);
    const bf16* We = expb + (size_t)l * 1632 * kD;
    k_gemm_qk   <<<dim3(kM / 128, 2),  128, 0, stream>>>(h, We, qhkh);
    k_gemm_geglu<<<dim3(kM / 128, 24), 128, 0, stream>>>(h, We, cat, attnvT);
    k_attn      <<<kM / 16, 128, 0, stream>>>(qhkh, attnvT, pbm + l, cat);
    k_gemm_proj <<<dim3(kM / 128, 6),  128, 0, stream>>>(cat, projb + (size_t)l * kD * kEX, x);
  }
  k_ln<<<kM / 8, 256, 0, stream>>>(x, ng, nb, h);
  k_gemm_logits<<<dim3(kM / 64, 2), 128, 0, stream>>>(h, outwb, out);
}
